// PCMerger_37134287241630
// MI455X (gfx1250) — matrix operand order repaired and checked by reading, not compiled, not run on hardware
//
#include <hip/hip_runtime.h>

// ---------------- problem constants (match reference) ----------------
#define BATCH   4
#define NPTS    100000
#define MPTS    100000
#define HID     128
#define KS_RAW  198          // sam input channels
#define KS_PAD  224          // padded to 7 * 32 for WMMA K-steps
#define XS_STRIDE 232        // bf16 elems per LDS row of staged sam input (464B, 16B-mult)
#define H_STRIDE  136        // bf16 elems per LDS row of hidden acts (272B, 16B-mult)
#define TILE_M  128
#define TILES_M ((MPTS + TILE_M - 1) / TILE_M)
#define TILE_N  64
#define TILES_N ((NPTS + TILE_N - 1) / TILE_N)
#define SV_STRIDE 132        // f32 elems per LDS row in gather staging (528B, 16B-mult)

typedef __attribute__((ext_vector_type(16))) __bf16 v16bf;
typedef __attribute__((ext_vector_type(8)))  float  v8f;

static __device__ __forceinline__ unsigned short f2bf(float f) {
  unsigned int u = __builtin_bit_cast(unsigned int, f);
  u += 0x7FFFu + ((u >> 16) & 1u);            // round-to-nearest-even
  return (unsigned short)(u >> 16);
}

static __device__ __forceinline__ v8f vzero8() {
  v8f z = {0.f, 0.f, 0.f, 0.f, 0.f, 0.f, 0.f, 0.f};
  return z;
}

// A fragment (16x32 bf16): lane holds row (lane&15); K groups at +8*(lane>=16) and +16.
static __device__ __forceinline__ v16bf ldfragA(const __bf16* p) {
  v16bf r;
  ((uint4*)&r)[0] = *(const uint4*)(p);       // K group 0..7
  ((uint4*)&r)[1] = *(const uint4*)(p + 16);  // K group 16..23
  return r;
}
// B fragment (32x16 bf16): lane holds column (lane&15); K groups at +8*(lane>=16) and +16.
static __device__ __forceinline__ v16bf ldfragB(const __bf16* p) {
  v16bf r;
  ((uint4*)&r)[0] = *(const uint4*)(p);
  ((uint4*)&r)[1] = *(const uint4*)(p + 16);
  return r;
}

// ---------------- kernel 0: weights f32 -> bf16 (padded) ----------------
__global__ void prep_weights(const float* __restrict__ s_w1,
                             const float* __restrict__ n_w2,
                             const float* __restrict__ r_w2,
                             const float* __restrict__ s_w2,
                             __bf16* __restrict__ w1s, __bf16* __restrict__ w2n,
                             __bf16* __restrict__ w2r, __bf16* __restrict__ w2s) {
  int gid = blockIdx.x * blockDim.x + threadIdx.x;
  if (gid < HID * KS_PAD) {
    int r = gid / KS_PAD, k = gid - r * KS_PAD;
    ((unsigned short*)w1s)[gid] = (k < KS_RAW) ? f2bf(s_w1[r * KS_RAW + k]) : (unsigned short)0;
  } else {
    int q = gid - HID * KS_PAD;
    if (q < 3 * HID * HID) {
      int mat = q / (HID * HID), e = q - mat * (HID * HID);
      const float* src = (mat == 0) ? n_w2 : (mat == 1) ? r_w2 : s_w2;
      __bf16*      dst = (mat == 0) ? w2n  : (mat == 1) ? w2r  : w2s;
      ((unsigned short*)dst)[e] = f2bf(src[e]);
    }
  }
}

// ---------------- kernel 1: fused 3-MLP tile -> combined[b][m][128] ----------------
__global__ __launch_bounds__(256) void fused_mlp_combined(
    const float* __restrict__ mv_feat,
    const float* __restrict__ n_w1, const float* __restrict__ n_b1, const float* __restrict__ n_b2,
    const float* __restrict__ r_w1, const float* __restrict__ r_b1, const float* __restrict__ r_b2,
    const float* __restrict__ s_b1, const float* __restrict__ s_b2,
    const __bf16* __restrict__ w1s, const __bf16* __restrict__ w2n,
    const __bf16* __restrict__ w2r, const __bf16* __restrict__ w2s,
    float* __restrict__ combined) {
  extern __shared__ char smem[];
  __bf16* hn = (__bf16*)smem;                       // [128][H_STRIDE]
  __bf16* hr = hn + TILE_M * H_STRIDE;              // [128][H_STRIDE]
  __bf16* xs = hr + TILE_M * H_STRIDE;              // [128][XS_STRIDE]
  __bf16* hs = xs;                                  // aliases xs (barrier-separated)

  const int tid  = threadIdx.x;
  const int lane = tid & 31;
  const int wv   = tid >> 5;
  const int tile = blockIdx.x % TILES_M;
  const int b    = blockIdx.x / TILES_M;
  const int m0   = tile * TILE_M;

  // ---- Phase A: K=3 first layers (normal / rgb) on VALU -> hn, hr (bf16) ----
  {
    const int rep = tid >> 7;                       // 0 -> normal, 1 -> rgb
    const int pm  = tid & 127;
    const int mg  = m0 + pm;
    const float* W1 = rep ? r_w1 : n_w1;
    const float* B1 = rep ? r_b1 : n_b1;
    __bf16* dst = rep ? hr : hn;
    float x0 = 0.f, x1 = 0.f, x2 = 0.f;
    if (mg < MPTS) {
      const float* mp = mv_feat + ((size_t)b * 204 + rep * 3) * MPTS + mg;
      x0 = mp[0]; x1 = mp[MPTS]; x2 = mp[2 * (size_t)MPTS];
    }
#pragma unroll 4
    for (int c = 0; c < HID; c += 2) {
      float f0 = fmaxf(fmaf(W1[c * 3 + 0], x0, fmaf(W1[c * 3 + 1], x1, fmaf(W1[c * 3 + 2], x2, B1[c]))), 0.f);
      float f1 = fmaxf(fmaf(W1[c * 3 + 3], x0, fmaf(W1[c * 3 + 4], x1, fmaf(W1[c * 3 + 5], x2, B1[c + 1]))), 0.f);
      *(unsigned int*)(dst + (size_t)pm * H_STRIDE + c) =
          (unsigned int)f2bf(f0) | ((unsigned int)f2bf(f1) << 16);
    }
  }

  // ---- Phase B: stage sam input tile transposed -> xs[pt][ch] (bf16, zero-padded) ----
  {
    const int pm = tid & 127;
    const int mg = m0 + pm;
    for (int cp = (tid >> 7); cp < KS_PAD / 2; cp += 2) {
      int ch = cp * 2;
      float v0 = 0.f, v1 = 0.f;
      if (mg < MPTS) {
        if (ch < KS_RAW)     v0 = mv_feat[((size_t)b * 204 + 6 + ch) * MPTS + mg];
        if (ch + 1 < KS_RAW) v1 = mv_feat[((size_t)b * 204 + 7 + ch) * MPTS + mg];
      }
      *(unsigned int*)(xs + (size_t)pm * XS_STRIDE + ch) =
          (unsigned int)f2bf(v0) | ((unsigned int)f2bf(v1) << 16);
    }
  }
  __syncthreads();

  const int cblk = wv * 16;                 // 16 output channels per wave
  const int arow = cblk + (lane & 15);      // A-matrix row for this lane
  const int ahi  = (lane >> 4) & 1;         // K-half selector

  // ---- Phase C: hs = relu(W1s[128x224] @ xs^T + b1s) via v_wmma_f32_16x16x32_bf16 ----
  v8f accs[8];
#pragma unroll
  for (int t = 0; t < 8; ++t) accs[t] = vzero8();
#pragma unroll
  for (int ks = 0; ks < KS_PAD / 32; ++ks) {
    const int kk = ks * 32;
    v16bf afr = ldfragA(w1s + (size_t)arow * KS_PAD + kk + ahi * 8);
    const int k0 = kk + ahi * 8;
#pragma unroll
    for (int t = 0; t < 8; ++t) {
      v16bf bfr = ldfragB(xs + (size_t)(t * 16 + (lane & 15)) * XS_STRIDE + k0);
      accs[t] = __builtin_amdgcn_wmma_f32_16x16x32_bf16(
          false, afr, false, bfr, (short)0, accs[t], false, false);
    }
  }
  __syncthreads();   // all waves done reading xs before hs (alias) is written
  {
    const int c0 = cblk + ahi * 8;
    float bias[8];
#pragma unroll
    for (int j = 0; j < 8; ++j) bias[j] = s_b1[c0 + j];
#pragma unroll
    for (int t = 0; t < 8; ++t) {
      unsigned short h[8];
#pragma unroll
      for (int j = 0; j < 8; ++j) h[j] = f2bf(fmaxf(accs[t][j] + bias[j], 0.f));
      uint4 pk;
      pk.x = (unsigned)h[0] | ((unsigned)h[1] << 16);
      pk.y = (unsigned)h[2] | ((unsigned)h[3] << 16);
      pk.z = (unsigned)h[4] | ((unsigned)h[5] << 16);
      pk.w = (unsigned)h[6] | ((unsigned)h[7] << 16);
      *(uint4*)(hs + (size_t)(t * 16 + (lane & 15)) * H_STRIDE + c0) = pk;
    }
  }
  __syncthreads();

  // ---- Phase D: combined = sum_g relu(W2g[128x128] @ hg + b2g), 3 GEMMs K=128 ----
  v8f sum[8];
#pragma unroll
  for (int t = 0; t < 8; ++t) sum[t] = vzero8();
  for (int g = 0; g < 3; ++g) {
    const __bf16* Wg = (g == 0) ? w2n : (g == 1) ? w2r : w2s;
    const __bf16* Hg = (g == 0) ? hn  : (g == 1) ? hr  : hs;
    const float*  Bg = (g == 0) ? n_b2 : (g == 1) ? r_b2 : s_b2;
    v8f acc[8];
#pragma unroll
    for (int t = 0; t < 8; ++t) acc[t] = vzero8();
#pragma unroll
    for (int ks = 0; ks < 4; ++ks) {
      const int kk = ks * 32;
      v16bf afr = ldfragA(Wg + (size_t)arow * HID + kk + ahi * 8);
      const int k0 = kk + ahi * 8;
#pragma unroll
      for (int t = 0; t < 8; ++t) {
        v16bf bfr = ldfragB(Hg + (size_t)(t * 16 + (lane & 15)) * H_STRIDE + k0);
        acc[t] = __builtin_amdgcn_wmma_f32_16x16x32_bf16(
            false, afr, false, bfr, (short)0, acc[t], false, false);
      }
    }
    const int c0 = cblk + ahi * 8;
#pragma unroll
    for (int t = 0; t < 8; ++t)
#pragma unroll
      for (int j = 0; j < 8; ++j)
        sum[t][j] += fmaxf(acc[t][j] + Bg[c0 + j], 0.f);
  }

  // store combined[b][m][c] (point-major so gather reads 512B/point)
  {
    const int c0 = cblk + ahi * 8;
#pragma unroll
    for (int t = 0; t < 8; ++t) {
      int mg = m0 + t * 16 + (lane & 15);
      if (mg < MPTS) {
        float* op = combined + ((size_t)b * MPTS + mg) * HID + c0;
        *(float4*)op       = make_float4(sum[t][0], sum[t][1], sum[t][2], sum[t][3]);
        *(float4*)(op + 4) = make_float4(sum[t][4], sum[t][5], sum[t][6], sum[t][7]);
      }
    }
  }
}

// ---------------- kernel 2: masked gather + add (two-phase through LDS) ----------------
// Phase 1: one wave per point -> 512B fully-coalesced read of combined[b][m][:].
// Phase 2: coalesced-along-n writes of out = feat + staged.
__global__ __launch_bounds__(256) void gather_add(
    const float* __restrict__ feat, const long long* __restrict__ idx,
    const float* __restrict__ combined, float* __restrict__ out) {
  __shared__ int   sidx[TILE_N];
  __shared__ float sval[TILE_N * SV_STRIDE];        // 64 x 132 f32 = 33,792 B

  const int tile = blockIdx.x % TILES_N;
  const int b    = blockIdx.x / TILES_N;
  const int n0   = tile * TILE_N;
  const int tid  = threadIdx.x;
  const int lane = tid & 31;
  const int wv   = tid >> 5;

  if (tid < TILE_N) {
    int n = n0 + tid;
    sidx[tid] = (n < NPTS) ? (int)idx[(size_t)b * NPTS + n] : -1;
  }
  __syncthreads();

  // phase 1: 8 waves x 8 points each; lane reads float4 at c = 4*lane
#pragma unroll
  for (int i = 0; i < 8; ++i) {
    const int p = wv * 8 + i;
    const int m = sidx[p];
    float4 v = make_float4(0.f, 0.f, 0.f, 0.f);
    if (m >= 0)
      v = *(const float4*)(combined + ((size_t)b * MPTS + m) * HID + lane * 4);
    *(float4*)(sval + (size_t)p * SV_STRIDE + lane * 4) = v;
  }
  __syncthreads();

  // phase 2: lanes span consecutive n for a fixed channel -> coalesced global R/W
  const int nl = tid & 63;
  const int cg = tid >> 6;
  const int n  = n0 + nl;
  if (n < NPTS) {
#pragma unroll 8
    for (int c = cg; c < HID; c += 4) {
      size_t o = ((size_t)b * HID + c) * NPTS + n;
      out[o] = feat[o] + sval[(size_t)nl * SV_STRIDE + c];
    }
  }
}

// ---------------- host launcher ----------------
extern "C" void kernel_launch(void* const* d_in, const int* in_sizes, int n_in,
                              void* d_out, int out_size, void* d_ws, size_t ws_size,
                              hipStream_t stream) {
  const float*     feat    = (const float*)d_in[0];
  const float*     mv_feat = (const float*)d_in[1];
  const long long* idx     = (const long long*)d_in[2];
  const float* n_w1 = (const float*)d_in[3];
  const float* n_b1 = (const float*)d_in[4];
  const float* n_w2 = (const float*)d_in[5];
  const float* n_b2 = (const float*)d_in[6];
  const float* r_w1 = (const float*)d_in[7];
  const float* r_b1 = (const float*)d_in[8];
  const float* r_w2 = (const float*)d_in[9];
  const float* r_b2 = (const float*)d_in[10];
  const float* s_w1 = (const float*)d_in[11];
  const float* s_b1 = (const float*)d_in[12];
  const float* s_w2 = (const float*)d_in[13];
  const float* s_b2 = (const float*)d_in[14];
  float* out = (float*)d_out;

  // workspace layout: combined f32 [B][M][128], then bf16 weight copies (16B-aligned)
  char* ws = (char*)d_ws;
  float* combined = (float*)ws;
  size_t off = (size_t)BATCH * MPTS * HID * sizeof(float);   // 204,800,000 (16B-mult)
  __bf16* w1s = (__bf16*)(ws + off);  off += (size_t)HID * KS_PAD * 2;
  __bf16* w2n = (__bf16*)(ws + off);  off += (size_t)HID * HID * 2;
  __bf16* w2r = (__bf16*)(ws + off);  off += (size_t)HID * HID * 2;
  __bf16* w2s = (__bf16*)(ws + off);

  const int prep_work = HID * KS_PAD + 3 * HID * HID;
  prep_weights<<<(prep_work + 255) / 256, 256, 0, stream>>>(
      s_w1, n_w2, r_w2, s_w2, w1s, w2n, w2r, w2s);

  const size_t lds_bytes = (size_t)(2 * TILE_M * H_STRIDE + TILE_M * XS_STRIDE) * 2; // 129,024
  fused_mlp_combined<<<dim3(BATCH * TILES_M), 256, lds_bytes, stream>>>(
      mv_feat, n_w1, n_b1, n_b2, r_w1, r_b1, r_b2, s_b1, s_b2,
      w1s, w2n, w2r, w2s, combined);

  gather_add<<<dim3(BATCH * TILES_N), 256, 0, stream>>>(feat, idx, combined, out);
}
